// FlowPokeTransformer_24086176596130
// MI455X (gfx1250) — matrix operand order repaired and checked by reading, not compiled, not run on hardware
//
#include <hip/hip_runtime.h>
#include <hip/hip_bf16.h>

typedef __bf16 bf16;
typedef __attribute__((ext_vector_type(16))) __bf16 v16bf;
typedef __attribute__((ext_vector_type(8)))  float  v8f;
typedef __attribute__((ext_vector_type(4)))  unsigned int u32x4;
typedef __attribute__((ext_vector_type(4)))  int v4i;

#define GLOBAL_AS __attribute__((address_space(1)))
#define LDS_AS    __attribute__((address_space(3)))

union FragBF {
    v16bf v;
    u32x4 q[2];
};

__device__ __forceinline__ u32x4 ld16(const bf16* p) {
    return *reinterpret_cast<const u32x4*>(p);
}

__device__ __forceinline__ v8f vzero8() {
    v8f z = {0.f,0.f,0.f,0.f,0.f,0.f,0.f,0.f};
    return z;
}

__device__ __forceinline__ v8f wmma_bf16(v16bf a, v16bf b, v8f c) {
    return __builtin_amdgcn_wmma_f32_16x16x32_bf16(
        /*neg_a=*/false, a, /*neg_b=*/false, b,
        /*c_mod=*/(short)0, c, /*reuse_a=*/false, /*reuse_b=*/false);
}

// 16-byte async global->LDS copy (per-lane), tracked by ASYNCcnt.
__device__ __forceinline__ void async_cp16(bf16* lds_dst, const bf16* gsrc) {
#if __has_builtin(__builtin_amdgcn_global_load_async_to_lds_b128)
    __builtin_amdgcn_global_load_async_to_lds_b128(
        (GLOBAL_AS v4i*)(void*)gsrc, (LDS_AS v4i*)(void*)lds_dst, 0, 0);
#else
    unsigned laddr = (unsigned)(unsigned long long)(LDS_AS void*)lds_dst;
    unsigned long long gaddr = (unsigned long long)gsrc;
    asm volatile("global_load_async_to_lds_b128 %0, %1, off"
                 :: "v"(laddr), "v"(gaddr) : "memory");
#endif
}

// ---------------------------------------------------------------------------
// fp32 -> bf16 conversion (n multiple of 4)
// ---------------------------------------------------------------------------
struct bf16x4 { bf16 a, b, c, d; };

__global__ __launch_bounds__(256) void cvt_f32_bf16(
    const float* __restrict__ in, bf16* __restrict__ out, int n)
{
    int i = (blockIdx.x * 256 + threadIdx.x) * 4;
    if (i >= n) return;
    float4 v = *reinterpret_cast<const float4*>(in + i);
    bf16x4 o;
    o.a = (bf16)v.x; o.b = (bf16)v.y; o.c = (bf16)v.z; o.d = (bf16)v.w;
    *reinterpret_cast<bf16x4*>(out + i) = o;
}

// ---------------------------------------------------------------------------
// C[M,N] (f32) = A[M,K] (bf16, row-major) * B[N,K]^T (bf16, row-major)
// block = 128x128 tile, 256 threads = 8 waves, each wave 32x64 (2x4 WMMA tiles)
// ---------------------------------------------------------------------------
__global__ __launch_bounds__(256) void gemm_bf16_nt(
    const bf16* __restrict__ A, const bf16* __restrict__ B,
    float* __restrict__ C, int M, int N, int K)
{
    const int lane = threadIdx.x & 31;
    const int wave = threadIdx.x >> 5;
    const int ln   = lane & 15;
    const int hi   = (lane >> 4) & 1;
    const int wm   = wave >> 1;           // 0..3
    const int wn   = wave & 1;            // 0..1
    const int m0   = blockIdx.y * 128 + wm * 32;
    const int n0   = blockIdx.x * 128 + wn * 64;

    v8f acc[2][4];
    #pragma unroll
    for (int i = 0; i < 2; ++i)
        #pragma unroll
        for (int j = 0; j < 4; ++j)
            acc[i][j] = vzero8();

    for (int k0 = 0; k0 < K; k0 += 32) {
        FragBF a[2];
        #pragma unroll
        for (int mt = 0; mt < 2; ++mt) {
            const bf16* ap = A + (size_t)(m0 + mt*16 + ln) * K + k0 + (hi ? 8 : 0);
            a[mt].q[0] = ld16(ap);
            a[mt].q[1] = ld16(ap + 16);
        }
        FragBF b[4];
        #pragma unroll
        for (int nt = 0; nt < 4; ++nt) {
            const bf16* bp = B + (size_t)(n0 + nt*16 + ln) * K + k0 + (hi ? 8 : 0);
            b[nt].q[0] = ld16(bp);
            b[nt].q[1] = ld16(bp + 16);
        }
        #pragma unroll
        for (int mt = 0; mt < 2; ++mt)
            #pragma unroll
            for (int nt = 0; nt < 4; ++nt)
                acc[mt][nt] = wmma_bf16(a[mt].v, b[nt].v, acc[mt][nt]);
    }

    #pragma unroll
    for (int mt = 0; mt < 2; ++mt)
        #pragma unroll
        for (int nt = 0; nt < 4; ++nt)
            #pragma unroll
            for (int r = 0; r < 8; ++r) {
                int row = m0 + mt*16 + r + 8*hi;
                int col = n0 + nt*16 + ln;
                C[(size_t)row * N + col] = acc[mt][nt][r];
            }
}

// ---------------------------------------------------------------------------
// Fused cos-scale + axial RoPE.  qkv[l][3072] f32 -> q_bf/k_bf [H][L][64],
// vt_bf [H][64][L] (transposed via LDS for coalesced stores).
// Block = 256 threads = 16 tokens x 16 heads.
// ---------------------------------------------------------------------------
__global__ __launch_bounds__(256) void rope_kernel(
    const float* __restrict__ qkv, const float* __restrict__ pos,
    const float* __restrict__ scale, const float* __restrict__ freqs,
    bf16* __restrict__ qb, bf16* __restrict__ kb, bf16* __restrict__ vt)
{
    __shared__ bf16 vbuf[16][16][64];   // [l_local][h][d]  32 KiB

    int tid = blockIdx.x * 256 + threadIdx.x;
    int h = tid & 15;
    int l = tid >> 4;
    int l_loc = l & 15;
    int l0 = blockIdx.x * 16;

    float thx = pos[l*2 + 0] * 2.f - 1.f;   // pos x -> th_w (freqs[0])
    float thy = pos[l*2 + 1] * 2.f - 1.f;   // pos y -> th_h (freqs[1])

    float cs[32], sn[32];
    #pragma unroll
    for (int j = 0; j < 16; ++j) {          // theta[0..15] = th_h
        float th = thy * freqs[256 + h*16 + j];
        cs[j] = __cosf(th); sn[j] = __sinf(th);
    }
    #pragma unroll
    for (int j = 0; j < 16; ++j) {          // theta[16..31] = th_w
        float th = thx * freqs[h*16 + j];
        cs[16 + j] = __cosf(th); sn[16 + j] = __sinf(th);
    }

    float ssc = sqrtf(fabsf(scale[h]) + 1e-8f);
    const float* base = qkv + (size_t)l * 3072 + h * 64;

    #pragma unroll
    for (int which = 0; which < 2; ++which) {       // 0 = q, 1 = k
        const float* src = base + which * 1024;
        float v[64];
        float sq = 0.f;
        #pragma unroll
        for (int d = 0; d < 64; d += 4) {
            float4 t = *reinterpret_cast<const float4*>(src + d);
            v[d] = t.x; v[d+1] = t.y; v[d+2] = t.z; v[d+3] = t.w;
            sq += t.x*t.x + t.y*t.y + t.z*t.z + t.w*t.w;
        }
        float rs = ssc * rsqrtf(sq + 1e-8f);
        bf16* dst = (which == 0 ? qb : kb) + ((size_t)h * 4096 + l) * 64;
        #pragma unroll
        for (int j = 0; j < 32; ++j) {
            float x1 = v[j] * rs, x2 = v[j + 32] * rs;
            dst[j]      = (bf16)(x1 * cs[j] - x2 * sn[j]);
            dst[j + 32] = (bf16)(x2 * cs[j] + x1 * sn[j]);
        }
    }

    // V: stage in LDS, transpose, write 32B-contiguous runs of 16 tokens.
    const float* vsrc = base + 2048;
    #pragma unroll
    for (int d = 0; d < 64; d += 4) {
        float4 t = *reinterpret_cast<const float4*>(vsrc + d);
        vbuf[l_loc][h][d]   = (bf16)t.x;
        vbuf[l_loc][h][d+1] = (bf16)t.y;
        vbuf[l_loc][h][d+2] = (bf16)t.z;
        vbuf[l_loc][h][d+3] = (bf16)t.w;
    }
    __syncthreads();
    #pragma unroll
    for (int it = 0; it < 4; ++it) {
        int pid = threadIdx.x + it * 256;   // 0..1023 -> (h,d)
        int hh = pid >> 6;
        int dd = pid & 63;
        bf16 tmp[16];
        #pragma unroll
        for (int i = 0; i < 16; ++i) tmp[i] = vbuf[i][hh][dd];
        u32x4* dstp = reinterpret_cast<u32x4*>(vt + ((size_t)hh*64 + dd)*4096 + l0);
        dstp[0] = *reinterpret_cast<u32x4*>(&tmp[0]);
        dstp[1] = *reinterpret_cast<u32x4*>(&tmp[8]);
    }
}

// ---------------------------------------------------------------------------
// Flash attention over pokes (causal).  4 waves/block, 16 queries/wave,
// 32 keys/iteration.  K/V tiles double-buffered in LDS via async-to-LDS DMA,
// shared by all 4 waves.  Output rows 0..3071 of att[L][1024] (bf16).
// ---------------------------------------------------------------------------
__global__ __launch_bounds__(128) void attn_pokes(
    const bf16* __restrict__ qb, const bf16* __restrict__ kb,
    const bf16* __restrict__ vt, bf16* __restrict__ att)
{
    __shared__ bf16 Kt [2][32][72];   // [buf][key][dh]   pad->144B rows
    __shared__ bf16 Vts[2][64][40];   // [buf][dh][key]   pad->80B rows
    __shared__ bf16 Ps [4][16][40];   // per-wave P tile

    const int h    = blockIdx.y;
    const int lane = threadIdx.x & 31;
    const int wave = threadIdx.x >> 5;
    const int ln   = lane & 15;
    const int hi   = (lane >> 4) & 1;
    const int q0   = blockIdx.x * 64 + wave * 16;
    const int nblk = blockIdx.x * 2 + 2;       // key tiles needed by block
    const int myend = (q0 + 15) / 32;          // last tile with work for wave

    auto issue_tile = [&](int blk) {
        int buf = blk & 1;
        int kbase = blk * 32;
        #pragma unroll
        for (int i = 0; i < 2; ++i) {          // K tile: 32x64, 256 chunks
            int c = threadIdx.x + i * 128;
            int key = c >> 3, off = (c & 7) * 8;
            async_cp16(&Kt[buf][key][off],
                       kb + ((size_t)h*4096 + kbase + key)*64 + off);
        }
        #pragma unroll
        for (int i = 0; i < 2; ++i) {          // V tile: 64x32, 256 chunks
            int c = threadIdx.x + i * 128;
            int d = c >> 2, off = (c & 3) * 8;
            async_cp16(&Vts[buf][d][off],
                       vt + ((size_t)h*64 + d)*4096 + kbase + off);
        }
    };

    FragBF aq[2];
    {
        const bf16* qp = qb + ((size_t)h * 4096 + q0 + ln) * 64;
        #pragma unroll
        for (int ks = 0; ks < 2; ++ks) {
            const bf16* p = qp + ks*32 + (hi ? 8 : 0);
            aq[ks].q[0] = ld16(p);
            aq[ks].q[1] = ld16(p + 16);
        }
    }

    v8f o[4];
    #pragma unroll
    for (int t = 0; t < 4; ++t) o[t] = vzero8();
    float m[8], lsum[8];
    #pragma unroll
    for (int r = 0; r < 8; ++r) { m[r] = -3.0e38f; lsum[r] = 0.f; }

    issue_tile(0);
    issue_tile(1);                             // nblk >= 2 always

    for (int blk = 0; blk < nblk; ++blk) {
        const int buf = blk & 1;
        const int kbase = blk * 32;

        if (blk + 1 < nblk) asm volatile("s_wait_asynccnt 0x4" ::: "memory");
        else                asm volatile("s_wait_asynccnt 0x0" ::: "memory");
        __syncthreads();

        if (blk <= myend) {
            v8f s[2] = { vzero8(), vzero8() };
            #pragma unroll
            for (int nt = 0; nt < 2; ++nt) {
                const bf16* kp = &Kt[buf][nt*16 + ln][0];
                #pragma unroll
                for (int ks = 0; ks < 2; ++ks) {
                    FragBF bfr;
                    const bf16* p = kp + ks*32 + (hi ? 8 : 0);
                    bfr.q[0] = ld16(p); bfr.q[1] = ld16(p + 16);
                    s[nt] = wmma_bf16(aq[ks].v, bfr.v, s[nt]);
                }
            }

            // causal mask: key > query -> -inf
            #pragma unroll
            for (int nt = 0; nt < 2; ++nt) {
                int key = kbase + nt*16 + ln;
                #pragma unroll
                for (int r = 0; r < 8; ++r)
                    if (key > q0 + r + 8*hi) s[nt][r] = -3.0e38f;
            }

            float p0[8], p1[8];
            #pragma unroll
            for (int r = 0; r < 8; ++r) {
                float mt = fmaxf(s[0][r], s[1][r]);
                #pragma unroll
                for (int off = 8; off >= 1; off >>= 1)
                    mt = fmaxf(mt, __shfl_xor(mt, off, 32));
                float mn   = fmaxf(m[r], mt);
                float corr = __expf(m[r] - mn);
                p0[r] = __expf(s[0][r] - mn);
                p1[r] = __expf(s[1][r] - mn);
                float rs = p0[r] + p1[r];
                #pragma unroll
                for (int off = 8; off >= 1; off >>= 1)
                    rs += __shfl_xor(rs, off, 32);
                lsum[r] = lsum[r] * corr + rs;
                m[r] = mn;
                #pragma unroll
                for (int t = 0; t < 4; ++t) o[t][r] *= corr;
            }

            // P (C-layout f32) -> LDS -> A-layout bf16
            #pragma unroll
            for (int r = 0; r < 8; ++r) {
                Ps[wave][r + 8*hi][ln]      = (bf16)p0[r];
                Ps[wave][r + 8*hi][16 + ln] = (bf16)p1[r];
            }
            __asm__ volatile("s_wait_dscnt 0" ::: "memory");

            FragBF ap;
            {
                const bf16* p = &Ps[wave][ln][hi ? 8 : 0];
                ap.q[0] = ld16(p);
                ap.q[1] = ld16(p + 16);
            }

            #pragma unroll
            for (int t = 0; t < 4; ++t) {
                FragBF bv;
                const bf16* p = &Vts[buf][t*16 + ln][hi ? 8 : 0];
                bv.q[0] = ld16(p); bv.q[1] = ld16(p + 16);
                o[t] = wmma_bf16(ap.v, bv.v, o[t]);
            }
        }

        __syncthreads();
        if (blk + 2 < nblk) issue_tile(blk + 2);
    }

    #pragma unroll
    for (int r = 0; r < 8; ++r) {
        float inv = 1.0f / lsum[r];
        int row = q0 + r + 8*hi;
        #pragma unroll
        for (int t = 0; t < 4; ++t)
            att[(size_t)row * 1024 + h*64 + t*16 + ln] = (bf16)(o[t][r] * inv);
    }
}

// ---------------------------------------------------------------------------
// Flash attention for queries: all 3072 pokes (no mask) + diagonal self-term.
// Same async double-buffered K/V staging.  Output rows 3072..4095 of att.
// ---------------------------------------------------------------------------
__global__ __launch_bounds__(128) void attn_query(
    const bf16* __restrict__ qb, const bf16* __restrict__ kb,
    const bf16* __restrict__ vt, bf16* __restrict__ att)
{
    __shared__ bf16 Kt [2][32][72];
    __shared__ bf16 Vts[2][64][40];
    __shared__ bf16 Ps [4][16][40];

    const int h    = blockIdx.y;
    const int lane = threadIdx.x & 31;
    const int wave = threadIdx.x >> 5;
    const int ln   = lane & 15;
    const int hi   = (lane >> 4) & 1;
    const int q0   = 3072 + blockIdx.x * 64 + wave * 16;
    const int nblk = 96;

    auto issue_tile = [&](int blk) {
        int buf = blk & 1;
        int kbase = blk * 32;
        #pragma unroll
        for (int i = 0; i < 2; ++i) {
            int c = threadIdx.x + i * 128;
            int key = c >> 3, off = (c & 7) * 8;
            async_cp16(&Kt[buf][key][off],
                       kb + ((size_t)h*4096 + kbase + key)*64 + off);
        }
        #pragma unroll
        for (int i = 0; i < 2; ++i) {
            int c = threadIdx.x + i * 128;
            int d = c >> 2, off = (c & 3) * 8;
            async_cp16(&Vts[buf][d][off],
                       vt + ((size_t)h*64 + d)*4096 + kbase + off);
        }
    };

    FragBF aq[2];
    {
        const bf16* qp = qb + ((size_t)h * 4096 + q0 + ln) * 64;
        #pragma unroll
        for (int ks = 0; ks < 2; ++ks) {
            const bf16* p = qp + ks*32 + (hi ? 8 : 0);
            aq[ks].q[0] = ld16(p);
            aq[ks].q[1] = ld16(p + 16);
        }
    }

    v8f o[4];
    #pragma unroll
    for (int t = 0; t < 4; ++t) o[t] = vzero8();
    float m[8], lsum[8];
    #pragma unroll
    for (int r = 0; r < 8; ++r) { m[r] = -3.0e38f; lsum[r] = 0.f; }

    issue_tile(0);
    issue_tile(1);

    for (int blk = 0; blk < nblk; ++blk) {
        const int buf = blk & 1;

        if (blk + 1 < nblk) asm volatile("s_wait_asynccnt 0x4" ::: "memory");
        else                asm volatile("s_wait_asynccnt 0x0" ::: "memory");
        __syncthreads();

        v8f s[2] = { vzero8(), vzero8() };
        #pragma unroll
        for (int nt = 0; nt < 2; ++nt) {
            const bf16* kp = &Kt[buf][nt*16 + ln][0];
            #pragma unroll
            for (int ks = 0; ks < 2; ++ks) {
                FragBF bfr;
                const bf16* p = kp + ks*32 + (hi ? 8 : 0);
                bfr.q[0] = ld16(p); bfr.q[1] = ld16(p + 16);
                s[nt] = wmma_bf16(aq[ks].v, bfr.v, s[nt]);
            }
        }

        float p0[8], p1[8];
        #pragma unroll
        for (int r = 0; r < 8; ++r) {
            float mt = fmaxf(s[0][r], s[1][r]);
            #pragma unroll
            for (int off = 8; off >= 1; off >>= 1)
                mt = fmaxf(mt, __shfl_xor(mt, off, 32));
            float mn   = fmaxf(m[r], mt);
            float corr = __expf(m[r] - mn);
            p0[r] = __expf(s[0][r] - mn);
            p1[r] = __expf(s[1][r] - mn);
            float rs = p0[r] + p1[r];
            #pragma unroll
            for (int off = 8; off >= 1; off >>= 1)
                rs += __shfl_xor(rs, off, 32);
            lsum[r] = lsum[r] * corr + rs;
            m[r] = mn;
            #pragma unroll
            for (int t = 0; t < 4; ++t) o[t][r] *= corr;
        }

        #pragma unroll
        for (int r = 0; r < 8; ++r) {
            Ps[wave][r + 8*hi][ln]      = (bf16)p0[r];
            Ps[wave][r + 8*hi][16 + ln] = (bf16)p1[r];
        }
        __asm__ volatile("s_wait_dscnt 0" ::: "memory");

        FragBF ap;
        {
            const bf16* p = &Ps[wave][ln][hi ? 8 : 0];
            ap.q[0] = ld16(p);
            ap.q[1] = ld16(p + 16);
        }

        #pragma unroll
        for (int t = 0; t < 4; ++t) {
            FragBF bv;
            const bf16* p = &Vts[buf][t*16 + ln][hi ? 8 : 0];
            bv.q[0] = ld16(p); bv.q[1] = ld16(p + 16);
            o[t] = wmma_bf16(ap.v, bv.v, o[t]);
        }

        __syncthreads();
        if (blk + 2 < nblk) issue_tile(blk + 2);
    }

    // diagonal self-term: score = q.k of own token, value = own v
    {
        int tok = q0 + ln;
        const bf16* qp = qb + ((size_t)h * 4096 + tok) * 64;
        const bf16* kp = kb + ((size_t)h * 4096 + tok) * 64;
        float sself = 0.f;
        #pragma unroll
        for (int d = 0; d < 64; d += 8) {
            u32x4 a4 = ld16(qp + d);
            u32x4 b4 = ld16(kp + d);
            const bf16* av = reinterpret_cast<const bf16*>(&a4);
            const bf16* bv = reinterpret_cast<const bf16*>(&b4);
            #pragma unroll
            for (int j = 0; j < 8; ++j)
                sself += (float)av[j] * (float)bv[j];
        }
        #pragma unroll
        for (int r = 0; r < 8; ++r) {
            int rowr = r + 8*hi;
            float sr    = __shfl(sself, rowr, 32);
            float mn    = fmaxf(m[r], sr);
            float corr  = __expf(m[r] - mn);
            float pself = __expf(sr - mn);
            lsum[r] = lsum[r] * corr + pself;
            int tokr = q0 + rowr;
            #pragma unroll
            for (int t = 0; t < 4; ++t) {
                float vv = (float)vt[((size_t)h*64 + t*16 + ln) * 4096 + tokr];
                o[t][r] = o[t][r] * corr + pself * vv;
            }
            m[r] = mn;
        }
    }

    #pragma unroll
    for (int r = 0; r < 8; ++r) {
        float inv = 1.0f / lsum[r];
        int row = q0 + r + 8*hi;
        #pragma unroll
        for (int t = 0; t < 4; ++t)
            att[(size_t)row * 1024 + h*64 + t*16 + ln] = (bf16)(o[t][r] * inv);
    }
}

// ---------------------------------------------------------------------------
// host-side launch
// ---------------------------------------------------------------------------
extern "C" void kernel_launch(void* const* d_in, const int* in_sizes, int n_in,
                              void* d_out, int out_size, void* d_ws, size_t ws_size,
                              hipStream_t stream)
{
    (void)in_sizes; (void)n_in; (void)out_size; (void)ws_size;

    const float* x     = (const float*)d_in[0];   // (1,4096,1024)
    const float* pos   = (const float*)d_in[1];   // (1,4096,2)
    const float* qkv_w = (const float*)d_in[2];   // (3072,1024)
    const float* out_w = (const float*)d_in[3];   // (1024,1024)
    const float* scale = (const float*)d_in[4];   // (16,)
    const float* freqs = (const float*)d_in[5];   // (2,16,16)
    // d_in[6] = n_pokes (compile-time constant 3072)

    char* ws = (char*)d_ws;
    bf16*  x_bf    = (bf16*)(ws + 0);                         //  8 MiB
    bf16*  qkvw_bf = (bf16*)(ws + (8u  << 20));               //  6 MiB
    bf16*  outw_bf = (bf16*)(ws + (8u  << 20) + 6291456u);    //  2 MiB
    float* qkvf    = (float*)(ws + (16u << 20));              // 48 MiB
    bf16*  q_bf    = (bf16*)(ws + (64u << 20));               //  8 MiB
    bf16*  k_bf    = (bf16*)(ws + (72u << 20));               //  8 MiB
    bf16*  vt_bf   = (bf16*)(ws + (80u << 20));               //  8 MiB
    bf16*  att_bf  = (bf16*)(ws + (88u << 20));               //  8 MiB

    // 1) fp32 -> bf16 conversions
    cvt_f32_bf16<<<4096*1024/4/256, 256, 0, stream>>>(x,     x_bf,    4096*1024);
    cvt_f32_bf16<<<3072*1024/4/256, 256, 0, stream>>>(qkv_w, qkvw_bf, 3072*1024);
    cvt_f32_bf16<<<1024*1024/4/256, 256, 0, stream>>>(out_w, outw_bf, 1024*1024);

    // 2) qkv = x @ qkv_w^T   (M=4096, N=3072, K=1024)
    gemm_bf16_nt<<<dim3(3072/128, 4096/128), 256, 0, stream>>>(
        x_bf, qkvw_bf, qkvf, 4096, 3072, 1024);

    // 3) cos-scale + RoPE + head-major repack (+ V transpose via LDS)
    rope_kernel<<<4096*16/256, 256, 0, stream>>>(
        qkvf, pos, scale, freqs, q_bf, k_bf, vt_bf);

    // 4) causal flash attention over pokes (rows 0..3071)
    attn_pokes<<<dim3(3072/64, 16), 128, 0, stream>>>(q_bf, k_bf, vt_bf, att_bf);

    // 5) query attention: all pokes + self (rows 3072..4095)
    attn_query<<<dim3(1024/64, 16), 128, 0, stream>>>(q_bf, k_bf, vt_bf, att_bf);

    // 6) out = att @ out_w^T  (M=4096, N=1024, K=1024) -> fp32 d_out
    gemm_bf16_nt<<<dim3(1024/128, 4096/128), 256, 0, stream>>>(
        att_bf, outw_bf, (float*)d_out, 4096, 1024, 1024);
}
